// MultiHeadSelfAttention_1331439862278
// MI455X (gfx1250) — matrix operand order repaired and checked by reading, not compiled, not run on hardware
//
#include <hip/hip_runtime.h>
#include <hip/hip_bf16.h>

// ---------------------------------------------------------------------------
// MI455X (gfx1250) multi-head self-attention, bf16 WMMA pipeline, fp32 accum.
// B=8, S=1024, D=768, H=12, dh=64.
// ---------------------------------------------------------------------------

#define B_ 8
#define S_ 1024
#define D_ 768
#define H_ 12
#define DH_ 64
#define N3D_ 2304        // 3*D
#define M_ (B_ * S_)     // 8192

typedef __attribute__((ext_vector_type(16))) __bf16 v16bf;
typedef __attribute__((ext_vector_type(2)))  __bf16 bf16x2;
typedef __attribute__((ext_vector_type(8)))  float  v8f;

union Frag16 { v16bf v; __bf16 e[16]; bf16x2 p[8]; };
union FragC  { v8f   v; float  e[8]; };

// float -> bf16, round-to-nearest-even
static __device__ inline __bf16 f2bf(float f) {
    union { float f; unsigned u; } v; v.f = f;
    unsigned r = v.u + 0x7FFFu + ((v.u >> 16) & 1u);
    union { unsigned short s; __bf16 b; } o; o.s = (unsigned short)(r >> 16);
    return o.b;
}

static __device__ inline v8f wmma_bf16(v16bf a, v16bf b, v8f c) {
    return __builtin_amdgcn_wmma_f32_16x16x32_bf16(
        /*neg_a=*/false, a, /*neg_b=*/false, b,
        /*c_mod=*/(short)0, c, /*reuse_a=*/false, /*reuse_b=*/false);
}

// A fragment (16x32 bf16, M x K). Lane l: row = l&15, g = l>>4.
// VGPR j (j<4): K = g*8 + 2j{,+1}; VGPR j+4: K = 16 + g*8 + 2j{,+1}.
static __device__ inline v16bf load_a_frag(const __bf16* base, int lda, int lane) {
    int row = lane & 15;
    int g   = lane >> 4;
    const __bf16* p = base + row * lda + g * 8;
    Frag16 f;
#pragma unroll
    for (int j = 0; j < 4; ++j) {
        f.p[j]     = *(const bf16x2*)(p + 2 * j);
        f.p[j + 4] = *(const bf16x2*)(p + 16 + 2 * j);
    }
    return f.v;
}

// B fragment (32x16 bf16, K x N). Lane l: N = l&15, g = l>>4.
// Element j (j<8): K = g*8 + j; element j+8: K = 16 + g*8 + j.
static __device__ inline v16bf load_b_frag(const __bf16* base, int ldb, int lane) {
    const int g = lane >> 4;
    const __bf16* p = base + (lane & 15);
    Frag16 f;
#pragma unroll
    for (int j = 0; j < 8; ++j) {
        f.e[j]     = p[(g * 8 + j) * ldb];
        f.e[j + 8] = p[(16 + g * 8 + j) * ldb];
    }
    return f.v;
}

// ---- CDNA5 async global->LDS staging (ASYNCcnt-tracked DMA path) ----------
static __device__ inline unsigned lds_offset_of(const __bf16* p) {
    // LDS (addrspace 3) pointers are workgroup-relative byte offsets
    typedef __attribute__((address_space(3))) const __bf16 lds_bf16;
    return (unsigned)(unsigned long long)(lds_bf16*)p;
}

static __device__ inline void async_b128_to_lds(unsigned lds_off, const void* gptr) {
    unsigned long long ga = (unsigned long long)gptr;
    asm volatile("global_load_async_to_lds_b128 %0, %1, off"
                 :: "v"(lds_off), "v"(ga) : "memory");
}

static __device__ inline void wait_asynccnt0() {
    asm volatile("s_wait_asynccnt 0x0" ::: "memory");
}

// ---------------------------------------------------------------------------
// fp32 -> bf16 conversion (grid-stride)
// ---------------------------------------------------------------------------
__global__ void f32_to_bf16_kernel(const float* __restrict__ in,
                                   __bf16* __restrict__ out, int n) {
    for (int i = blockIdx.x * blockDim.x + threadIdx.x; i < n;
         i += gridDim.x * blockDim.x)
        out[i] = f2bf(in[i]);
}

// ---------------------------------------------------------------------------
// Phase 1: qkv = x @ Wqkv + bqkv ; scatter into Q[B,H,S,dh], Kt[B,H,dh,S],
// V[B,H,S,dh] (bf16). One wave per 32x64 output tile (2 A x 4 B frags,
// 8 WMMAs per 32-wide k-step).
// ---------------------------------------------------------------------------
__global__ __launch_bounds__(256) void qkv_gemm_kernel(
    const __bf16* __restrict__ xb, const __bf16* __restrict__ wqkvb,
    const float* __restrict__ bqkv,
    __bf16* __restrict__ Qb, __bf16* __restrict__ Ktb, __bf16* __restrict__ Vb) {
    const int lane = threadIdx.x & 31;
    const int wave = threadIdx.x >> 5;
    const int tile = blockIdx.x * 8 + wave;
    const int TN = N3D_ / 64;                 // 36
    const int tm = tile / TN, tn = tile % TN;
    const int m0 = tm * 32, n0 = tn * 64;

    FragC acc[4][2];
#pragma unroll
    for (int t = 0; t < 4; ++t)
#pragma unroll
        for (int r = 0; r < 2; ++r) { v8f z = {}; acc[t][r].v = z; }

#pragma unroll 2
    for (int k0 = 0; k0 < D_; k0 += 32) {
        v16bf a0 = load_a_frag(xb + (size_t)m0 * D_ + k0, D_, lane);
        v16bf a1 = load_a_frag(xb + (size_t)(m0 + 16) * D_ + k0, D_, lane);
#pragma unroll
        for (int t = 0; t < 4; ++t) {
            v16bf b = load_b_frag(wqkvb + (size_t)k0 * N3D_ + n0 + t * 16, N3D_, lane);
            acc[t][0].v = wmma_bf16(a0, b, acc[t][0].v);
            acc[t][1].v = wmma_bf16(a1, b, acc[t][1].v);
        }
    }

    const int g = lane >> 4, col = lane & 15;
#pragma unroll
    for (int t = 0; t < 4; ++t) {
        const int nn = n0 + t * 16 + col;
        const int which = nn / D_;            // 0=q 1=k 2=v
        const int dcol  = nn % D_;
        const int h  = dcol >> 6;
        const int dd = dcol & 63;
        const float bv = bqkv[nn];
#pragma unroll
        for (int r = 0; r < 2; ++r) {
#pragma unroll
            for (int i = 0; i < 8; ++i) {
                const int m = m0 + r * 16 + i + 8 * g;
                const int b_ = m >> 10, s = m & 1023;
                const __bf16 o = f2bf(acc[t][r].e[i] + bv);
                if (which == 0)
                    Qb[(((size_t)(b_ * H_ + h)) * S_ + s) * DH_ + dd] = o;
                else if (which == 1)
                    Ktb[(((size_t)(b_ * H_ + h)) * DH_ + dd) * S_ + s] = o; // transposed
                else
                    Vb[(((size_t)(b_ * H_ + h)) * S_ + s) * DH_ + dd] = o;
            }
        }
    }
}

// ---------------------------------------------------------------------------
// Phase 2: flash-style attention. 8 waves/block share one (b,h); K/V tiles
// staged via async global->LDS DMA (ASYNCcnt). Online softmax in fp32;
// P staged through per-wave LDS to convert C-layout -> A-layout.
// ---------------------------------------------------------------------------
__global__ __launch_bounds__(256) void attn_kernel(
    const __bf16* __restrict__ Qb, const __bf16* __restrict__ Ktb,
    const __bf16* __restrict__ Vb, __bf16* __restrict__ ctxb) {
    __shared__ __bf16 lds_k[64 * 32];         // K^T slice: [dh=64][32 keys]
    __shared__ __bf16 lds_v[32 * 64];         // V slice:   [32 keys][dh=64]
    __shared__ __bf16 pstage[8][16 * 32];     // per-wave P staging

    const int tid  = threadIdx.x;
    const int lane = tid & 31;
    const int wave = tid >> 5;
    const int bh = blockIdx.x >> 3;           // 96 (b,h) pairs
    const int qgroup = blockIdx.x & 7;
    const int b = bh / H_, h = bh % H_;
    const int q0 = (qgroup * 8 + wave) * 16;

    const __bf16* Q  = Qb  + ((size_t)(b * H_ + h)) * S_ * DH_;
    const __bf16* Kt = Ktb + ((size_t)(b * H_ + h)) * DH_ * S_;
    const __bf16* V  = Vb  + ((size_t)(b * H_ + h)) * S_ * DH_;

    // Per-thread async staging slots (loop-invariant LDS offsets)
    const int krow = tid >> 2, kc = (tid & 3) * 8;   // K^T: 64 rows x 32 keys
    const int vrow = tid >> 3, vc = (tid & 7) * 8;   // V:   32 rows x 64 d
    const unsigned k_lds = lds_offset_of(&lds_k[krow * 32 + kc]);
    const unsigned v_lds = lds_offset_of(&lds_v[vrow * 64 + vc]);

    // Q tile 16x64 resident in registers as two A fragments (K=0..31, 32..63)
    const v16bf qa0 = load_a_frag(Q + (size_t)q0 * DH_ + 0,  DH_, lane);
    const v16bf qa1 = load_a_frag(Q + (size_t)q0 * DH_ + 32, DH_, lane);

    FragC ctx[4];
#pragma unroll
    for (int t = 0; t < 4; ++t) { v8f z = {}; ctx[t].v = z; }
    float mrow[8], lrow[8];
#pragma unroll
    for (int i = 0; i < 8; ++i) { mrow[i] = -1e30f; lrow[i] = 0.0f; }

    __bf16* pw = pstage[wave];
    const int g = lane >> 4, col = lane & 15;

    for (int key0 = 0; key0 < S_; key0 += 32) {
        __syncthreads();                      // previous readers done
        // async DMA: memory -> LDS, no VGPR round-trip, tracked by ASYNCcnt
        async_b128_to_lds(k_lds, &Kt[(size_t)krow * S_ + key0 + kc]);
        async_b128_to_lds(v_lds, &V[(size_t)(key0 + vrow) * DH_ + vc]);
        wait_asynccnt0();
        __syncthreads();

        // scores for two 16-key tiles: S = Q(16x64) @ K^T(64x16) each
        v8f s0 = {}, s1 = {};
        {
            v16bf kb;
            kb = load_b_frag(lds_k + 0 * 32 + 0,  32, lane); s0 = wmma_bf16(qa0, kb, s0);
            kb = load_b_frag(lds_k + 32 * 32 + 0, 32, lane); s0 = wmma_bf16(qa1, kb, s0);
            kb = load_b_frag(lds_k + 0 * 32 + 16, 32, lane); s1 = wmma_bf16(qa0, kb, s1);
            kb = load_b_frag(lds_k + 32 * 32 + 16,32, lane); s1 = wmma_bf16(qa1, kb, s1);
        }
        FragC f0, f1; f0.v = s0; f1.v = s1;

        // online softmax, rows r = i + 8*g, key columns spread over 16 lanes
#pragma unroll
        for (int i = 0; i < 8; ++i) {
            float a = f0.e[i] * 0.125f;       // 1/sqrt(64)
            float c2 = f1.e[i] * 0.125f;
            float tmax = fmaxf(a, c2);
#pragma unroll
            for (int off = 1; off < 16; off <<= 1)
                tmax = fmaxf(tmax, __shfl_xor(tmax, off, 32));
            const float mnew  = fmaxf(mrow[i], tmax);
            const float scale = __expf(mrow[i] - mnew);
            const float e0 = __expf(a - mnew);
            const float e1 = __expf(c2 - mnew);
            float rs = e0 + e1;
#pragma unroll
            for (int off = 1; off < 16; off <<= 1)
                rs += __shfl_xor(rs, off, 32);
            lrow[i] = lrow[i] * scale + rs;
            mrow[i] = mnew;
            const int r = i + 8 * g;
            pw[r * 32 + col]      = f2bf(e0);
            pw[r * 32 + 16 + col] = f2bf(e1);
#pragma unroll
            for (int t = 0; t < 4; ++t) ctx[t].e[i] *= scale;
        }

        __builtin_amdgcn_wave_barrier();      // keep ds stores before ds loads

        // P (16x32) back as A fragment; ctx += P @ V(32x64)
        const v16bf pa = load_a_frag(pw, 32, lane);
#pragma unroll
        for (int t = 0; t < 4; ++t) {
            v16bf vb = load_b_frag(lds_v + t * 16, 64, lane);
            ctx[t].v = wmma_bf16(pa, vb, ctx[t].v);
        }
    }

    // epilogue: ctx / l  ->  ctxb [B,S,D] bf16
#pragma unroll
    for (int t = 0; t < 4; ++t) {
#pragma unroll
        for (int i = 0; i < 8; ++i) {
            const int r = i + 8 * g;
            const float o = ctx[t].e[i] / lrow[i];
            ctxb[((size_t)b * S_ + q0 + r) * D_ + h * DH_ + t * 16 + col] = f2bf(o);
        }
    }
}

// ---------------------------------------------------------------------------
// Phase 3: out = ctx @ Wo + bo (fp32 output); 32x64 tile per wave.
// ---------------------------------------------------------------------------
__global__ __launch_bounds__(256) void out_gemm_kernel(
    const __bf16* __restrict__ ctxb, const __bf16* __restrict__ wob,
    const float* __restrict__ bo, float* __restrict__ out) {
    const int lane = threadIdx.x & 31;
    const int wave = threadIdx.x >> 5;
    const int tile = blockIdx.x * 8 + wave;
    const int TN = D_ / 64;                   // 12
    const int tm = tile / TN, tn = tile % TN;
    const int m0 = tm * 32, n0 = tn * 64;

    FragC acc[4][2];
#pragma unroll
    for (int t = 0; t < 4; ++t)
#pragma unroll
        for (int r = 0; r < 2; ++r) { v8f z = {}; acc[t][r].v = z; }

#pragma unroll 2
    for (int k0 = 0; k0 < D_; k0 += 32) {
        v16bf a0 = load_a_frag(ctxb + (size_t)m0 * D_ + k0, D_, lane);
        v16bf a1 = load_a_frag(ctxb + (size_t)(m0 + 16) * D_ + k0, D_, lane);
#pragma unroll
        for (int t = 0; t < 4; ++t) {
            v16bf b = load_b_frag(wob + (size_t)k0 * D_ + n0 + t * 16, D_, lane);
            acc[t][0].v = wmma_bf16(a0, b, acc[t][0].v);
            acc[t][1].v = wmma_bf16(a1, b, acc[t][1].v);
        }
    }

    const int g = lane >> 4, col = lane & 15;
#pragma unroll
    for (int t = 0; t < 4; ++t) {
        const int nn = n0 + t * 16 + col;
        const float bv = bo[nn];
#pragma unroll
        for (int r = 0; r < 2; ++r) {
#pragma unroll
            for (int i = 0; i < 8; ++i) {
                const int m = m0 + r * 16 + i + 8 * g;
                out[(size_t)m * D_ + nn] = acc[t][r].e[i] + bv;
            }
        }
    }
}

// ---------------------------------------------------------------------------
// Launch
// ---------------------------------------------------------------------------
extern "C" void kernel_launch(void* const* d_in, const int* in_sizes, int n_in,
                              void* d_out, int out_size, void* d_ws, size_t ws_size,
                              hipStream_t stream) {
    (void)in_sizes; (void)n_in; (void)out_size; (void)ws_size;
    const float* x    = (const float*)d_in[0];   // [8,1024,768]
    const float* Wqkv = (const float*)d_in[1];   // [768,2304]
    const float* bqkv = (const float*)d_in[2];   // [2304]
    const float* Wo   = (const float*)d_in[3];   // [768,768]
    const float* bo   = (const float*)d_in[4];   // [768]
    float* out = (float*)d_out;

    // workspace partition (bf16 buffers)
    char* w = (char*)d_ws;
    __bf16* xb    = (__bf16*)(w);                                   // 12.6 MB
    __bf16* wqkvb = (__bf16*)(w + 12582912);                        //  3.5 MB
    __bf16* wob   = (__bf16*)(w + 12582912 + 3538944);              //  1.2 MB
    __bf16* Qb    = (__bf16*)(w + 17301504);                        // 12.6 MB
    __bf16* Ktb   = (__bf16*)(w + 17301504 + 12582912);             // 12.6 MB
    __bf16* Vb    = (__bf16*)(w + 17301504 + 2 * 12582912);         // 12.6 MB
    __bf16* ctxb  = (__bf16*)(w + 17301504 + 3 * 12582912);         // 12.6 MB

    // fp32 -> bf16 conversions
    const int nx = M_ * D_;           // 6291456
    const int nw = D_ * N3D_;         // 1769472
    const int no = D_ * D_;           // 589824
    f32_to_bf16_kernel<<<(nx + 1023) / 1024, 256, 0, stream>>>(x, xb, nx);
    f32_to_bf16_kernel<<<(nw + 1023) / 1024, 256, 0, stream>>>(Wqkv, wqkvb, nw);
    f32_to_bf16_kernel<<<(no + 1023) / 1024, 256, 0, stream>>>(Wo, wob, no);

    // QKV projection: 256 x 36 (32x64) tiles, 8 waves/block
    qkv_gemm_kernel<<<(M_ / 32) * (N3D_ / 64) / 8, 256, 0, stream>>>(
        xb, wqkvb, bqkv, Qb, Ktb, Vb);

    // attention: 96 (b,h) x 8 q-groups
    attn_kernel<<<B_ * H_ * 8, 256, 0, stream>>>(Qb, Ktb, Vb, ctxb);

    // output projection: 256 x 12 (32x64) tiles, 8 waves/block
    out_gemm_kernel<<<(M_ / 32) * (D_ / 64) / 8, 256, 0, stream>>>(
        ctxb, wob, bo, out);
}
